// Model_19241453486294
// MI455X (gfx1250) — matrix operand order repaired and checked by reading, not compiled, not run on hardware
//
#include <hip/hip_runtime.h>

#define NN 128
#define CC 128
#define TT 64
#define VV 25
#define HH 4
#define HD 32
#define TV 1600        // T*V
#define CTV 204800     // C*T*V
#define NCTV 26214400  // N*C*T*V

typedef __attribute__((ext_vector_type(16))) __bf16 v16bf;
typedef __attribute__((ext_vector_type(8)))  float  v8f;

union V16U  { v16bf v; uint4 q[2]; unsigned short s[16]; };
union U16x8 { uint4 q; unsigned short s[8]; };

static __device__ __forceinline__ unsigned short f2bf(float f) {
  union { float f; unsigned int u; } c; c.f = f;
  unsigned int u = c.u;
  u += 0x7FFFu + ((u >> 16) & 1u);            // round-to-nearest-even
  return (unsigned short)(u >> 16);
}
static __device__ __forceinline__ float bf2f(unsigned short s) {
  union { unsigned int u; float f; } c; c.u = ((unsigned int)s) << 16;
  return c.f;
}
static __device__ __forceinline__ v8f wmma_bf16(v16bf a, v16bf b, v8f c) {
  return __builtin_amdgcn_wmma_f32_16x16x32_bf16(false, a, false, b, (short)0, c, false, false);
}

// A fragment (16x32, K contiguous) from row-major src[m][k]; 2 x 16B vector loads.
static __device__ __forceinline__ v16bf load_A_rm(const unsigned short* src, int ld, int lane) {
  int row = lane & 15;
  int ko  = (lane >> 4) << 3;
  const unsigned short* p = src + row * ld + ko;
  V16U r;
  r.q[0] = *(const uint4*)(p);
  r.q[1] = *(const uint4*)(p + 16);
  return r.v;
}
// A fragment with per-lane row predication (zero fill out of range rows)
static __device__ __forceinline__ v16bf load_A_rm_pred(const unsigned short* srcRow0, int ld,
                                                       int lane, int rowBase, int rowMax) {
  int row = rowBase + (lane & 15);
  int ko  = (lane >> 4) << 3;
  V16U r;
  if (row >= 0 && row < rowMax) {
    const unsigned short* p = srcRow0 + row * ld + ko;
    r.q[0] = *(const uint4*)(p);
    r.q[1] = *(const uint4*)(p + 16);
  } else {
    r.q[0] = make_uint4(0u, 0u, 0u, 0u);
    r.q[1] = make_uint4(0u, 0u, 0u, 0u);
  }
  return r.v;
}
// B fragment (32x16) from column-major source: B[k][n] = src[n][k]; 2 x 16B vector loads.
static __device__ __forceinline__ v16bf load_B_cm(const unsigned short* src, int ld, int lane) {
  int col = lane & 15;
  int kb  = (lane >> 4) << 3;
  const unsigned short* p = src + col * ld + kb;
  V16U r;
  r.q[0] = *(const uint4*)(p);
  r.q[1] = *(const uint4*)(p + 16);
  return r.v;
}
// A fragment from transposed source: A[m][k] = src[k][m] (scalar gather; small uses only)
static __device__ __forceinline__ v16bf load_At(const unsigned short* src, int ld, int lane) {
  int row = lane & 15;
  int ko  = (lane >> 4) << 3;
  V16U r;
#pragma unroll
  for (int i = 0; i < 8; ++i) r.s[i]     = src[(ko + i) * ld + row];
#pragma unroll
  for (int i = 0; i < 8; ++i) r.s[8 + i] = src[(16 + ko + i) * ld + row];
  return r.v;
}

// ---------------- transposes / repacks ----------------
// x[n][c][col] f32 -> xbT[n][col][c] bf16
__global__ void __launch_bounds__(128) k_tr1(const float* __restrict__ x, unsigned short* __restrict__ xbT) {
  __shared__ unsigned short Ls[64 * 136];
  const int n = blockIdx.y, col0 = blockIdx.x * 64, tid = threadIdx.x;
  for (int i = tid; i < 8192; i += 128) {
    int c = i >> 6, j = i & 63;
    Ls[j * 136 + c] = f2bf(x[n * CTV + c * TV + col0 + j]);
  }
  __syncthreads();
  for (int i = tid; i < 8192; i += 128) {
    int j = i >> 7, c = i & 127;
    xbT[n * CTV + (col0 + j) * 128 + c] = Ls[j * 136 + c];
  }
}
// x[n][c][t][v] f32 -> xbT2[n][v][c][t] bf16
__global__ void __launch_bounds__(128) k_tr2(const float* __restrict__ x, unsigned short* __restrict__ xbT2) {
  __shared__ unsigned short Ls[1600];
  const int n = blockIdx.y, c = blockIdx.x, tid = threadIdx.x;
  const int base = n * CTV + c * TV;
  for (int i = tid; i < 1600; i += 128) {
    int t = i / 25, v = i - t * 25;
    Ls[v * 64 + t] = f2bf(x[base + i]);
  }
  __syncthreads();
  for (int i = tid; i < 1600; i += 128) {
    int v = i >> 6, t = i & 63;
    xbT2[((n * 25 + v) * 128 + c) * 64 + t] = Ls[i];
  }
}
__global__ void k_cast(const float* __restrict__ src, unsigned short* __restrict__ dst, int n) {
  int i = blockIdx.x * blockDim.x + threadIdx.x;
  int st = gridDim.x * blockDim.x;
  for (; i < n; i += st) dst[i] = f2bf(src[i]);
}
// Tm[v][t][q] -> TmT[v][q][t]
__global__ void k_tmT(const float* __restrict__ Tm, unsigned short* __restrict__ TmT) {
  int i = blockIdx.x * blockDim.x + threadIdx.x;
  if (i >= 25 * 4096) return;
  int v = i >> 12, r = i & 4095, q = r >> 6, t = r & 63;
  TmT[i] = f2bf(Tm[v * 4096 + t * 64 + q]);
}
// A[t][v][w] -> aT[t][w][v] padded to 32x32
__global__ void k_aT(const float* __restrict__ A, unsigned short* __restrict__ aT) {
  int i = blockIdx.x * blockDim.x + threadIdx.x;
  if (i >= TT * 1024) return;
  int t = i >> 10, r = i & 1023, w = r >> 5, v = r & 31;
  float val = (v < VV && w < VV) ? A[(t * VV + v) * VV + w] : 0.f;
  aT[i] = f2bf(val);
}
// w_tcn[o][c][dt][0] -> wt3[dt][o][c]
__global__ void k_wtcn(const float* __restrict__ w, unsigned short* __restrict__ wt3) {
  int i = blockIdx.x * blockDim.x + threadIdx.x;
  if (i >= 3 * CC * CC) return;
  int dt = i >> 14, r = i & 16383, o = r >> 7, c = r & 127;
  wt3[i] = f2bf(w[(o * CC + c) * 3 + dt]);
}

// ---------------- gc1: per (n,v): Y^T[c][q] = X^T[c][t] x Tm_v[t][q]; out y1c[n][q][c][v32] ----------------
__global__ void __launch_bounds__(128) k_gc1(const unsigned short* __restrict__ xbT2,
                                             const unsigned short* __restrict__ TmT,
                                             unsigned short* __restrict__ y1c) {
  const int v = blockIdx.x;   // 0..31 (25..31 => zero-fill pad lanes of K for gc2)
  const int n = blockIdx.y;
  const int tid = threadIdx.x, lane = tid & 31, wid = tid >> 5;
  if (v >= VV) {
    for (int i = tid; i < 8192; i += 128) {            // q(64) x c(128)
      int q = i >> 7, c = i & 127;
      y1c[((n * 64 + q) * 128 + c) * 32 + v] = 0;
    }
    return;
  }
  const unsigned short* Xa = xbT2 + ((n * 25 + v) * 128) * 64;   // [c][t]
  const unsigned short* Tb = TmT + v * 4096;                     // [q][t]
  const int r0 = wid * 32;
  v8f z = {};
  v8f acc[2][4];
#pragma unroll
  for (int mt = 0; mt < 2; ++mt)
#pragma unroll
    for (int nt = 0; nt < 4; ++nt) acc[mt][nt] = z;
#pragma unroll
  for (int ks = 0; ks < 64; ks += 32) {
    v16bf a0 = load_A_rm(Xa + (r0) * 64 + ks, 64, lane);
    v16bf a1 = load_A_rm(Xa + (r0 + 16) * 64 + ks, 64, lane);
#pragma unroll
    for (int nt = 0; nt < 4; ++nt) {
      v16bf b = load_B_cm(Tb + (nt * 16) * 64 + ks, 64, lane);
      acc[0][nt] = wmma_bf16(a0, b, acc[0][nt]);
      acc[1][nt] = wmma_bf16(a1, b, acc[1][nt]);
    }
  }
  int Mo = (lane >> 4) << 3, nc = lane & 15;
#pragma unroll
  for (int mt = 0; mt < 2; ++mt)
#pragma unroll
    for (int nt = 0; nt < 4; ++nt) {
      int q = nt * 16 + nc;
#pragma unroll
      for (int i = 0; i < 8; ++i) {
        int c = r0 + mt * 16 + Mo + i;
        y1c[((n * 64 + q) * 128 + c) * 32 + v] = f2bf(acc[mt][nt][i]);
      }
    }
}

// ---------------- gc2: per (n,t): XG^T[c][w] = Y^T[c][v32] x A_t[v][w]; out xgT[n][t*25+w][c] ----------------
__global__ void __launch_bounds__(128) k_gc2(const unsigned short* __restrict__ y1c,
                                             const unsigned short* __restrict__ aT,
                                             unsigned short* __restrict__ xgT) {
  const int t = blockIdx.x;
  const int n = blockIdx.y;
  const int tid = threadIdx.x, lane = tid & 31, wid = tid >> 5;
  const unsigned short* Ya = y1c + ((n * 64 + t) * 128) * 32;    // [c][v32]
  const unsigned short* Ab = aT + t * 1024;                      // [w][v]
  const int r0 = wid * 32;
  v8f z = {};
  v8f acc[2][2];
#pragma unroll
  for (int mt = 0; mt < 2; ++mt)
#pragma unroll
    for (int nt = 0; nt < 2; ++nt) acc[mt][nt] = z;
  v16bf a0 = load_A_rm(Ya + (r0) * 32, 32, lane);
  v16bf a1 = load_A_rm(Ya + (r0 + 16) * 32, 32, lane);
#pragma unroll
  for (int nt = 0; nt < 2; ++nt) {
    v16bf b = load_B_cm(Ab + (nt * 16) * 32, 32, lane);
    acc[0][nt] = wmma_bf16(a0, b, acc[0][nt]);
    acc[1][nt] = wmma_bf16(a1, b, acc[1][nt]);
  }
  int Mo = (lane >> 4) << 3, nc = lane & 15;
#pragma unroll
  for (int mt = 0; mt < 2; ++mt)
#pragma unroll
    for (int nt = 0; nt < 2; ++nt) {
      int w = nt * 16 + nc;
      if (w < VV) {
        U16x8 u;
#pragma unroll
        for (int i = 0; i < 8; ++i) u.s[i] = f2bf(acc[mt][nt][i]);
        *(uint4*)(xgT + n * CTV + (t * VV + w) * 128 + r0 + mt * 16 + Mo) = u.q;
      }
    }
}

// ---------------- projections: OUT[o][col] = W[o][:] x X[:, col] + b; out transposed [n][col][o] ----------------
__global__ void __launch_bounds__(128) k_proj(const unsigned short* __restrict__ xbT,
    const unsigned short* __restrict__ wqb, const unsigned short* __restrict__ wkb,
    const unsigned short* __restrict__ wvb,
    const float* __restrict__ bq, const float* __restrict__ bk, const float* __restrict__ bv,
    unsigned short* __restrict__ qT, unsigned short* __restrict__ kT,
    unsigned short* __restrict__ vT) {
  const int col0 = blockIdx.x * 64;
  const int n = blockIdx.y;
  const int which = blockIdx.z;
  const unsigned short* W = (which == 0) ? wqb : (which == 1) ? wkb : wvb;
  const float* bias = (which == 0) ? bq : (which == 1) ? bk : bv;
  unsigned short* outT = (which == 0) ? qT : (which == 1) ? kT : vT;
  const int tid = threadIdx.x, lane = tid & 31, wid = tid >> 5;
  const int r0 = wid * 32;                               // o rows for this wave
  const unsigned short* Xn = xbT + n * CTV;              // [col][c]
  v8f z = {};
  v8f acc[2][4];
#pragma unroll
  for (int mt = 0; mt < 2; ++mt)
#pragma unroll
    for (int nt = 0; nt < 4; ++nt) acc[mt][nt] = z;
#pragma unroll
  for (int ks = 0; ks < 128; ks += 32) {
    v16bf a0 = load_A_rm(W + (r0) * 128 + ks, 128, lane);
    v16bf a1 = load_A_rm(W + (r0 + 16) * 128 + ks, 128, lane);
#pragma unroll
    for (int nt = 0; nt < 4; ++nt) {
      v16bf b = load_B_cm(Xn + (col0 + nt * 16) * 128 + ks, 128, lane);
      acc[0][nt] = wmma_bf16(a0, b, acc[0][nt]);
      acc[1][nt] = wmma_bf16(a1, b, acc[1][nt]);
    }
  }
  int Mo = (lane >> 4) << 3, nc = lane & 15;
#pragma unroll
  for (int mt = 0; mt < 2; ++mt) {
    int ob = r0 + mt * 16 + Mo;
    float4 b0 = *(const float4*)(bias + ob);
    float4 b1 = *(const float4*)(bias + ob + 4);
    float bv8[8] = {b0.x, b0.y, b0.z, b0.w, b1.x, b1.y, b1.z, b1.w};
#pragma unroll
    for (int nt = 0; nt < 4; ++nt) {
      int col = col0 + nt * 16 + nc;
      U16x8 u;
#pragma unroll
      for (int i = 0; i < 8; ++i) u.s[i] = f2bf(acc[mt][nt][i] + bv8[i]);
      *(uint4*)(outT + n * CTV + col * 128 + ob) = u.q;
    }
  }
}

// ---------------- attention per (n,h,t) ----------------
__global__ void __launch_bounds__(32) k_attn(const unsigned short* __restrict__ qT,
    const unsigned short* __restrict__ kT, const unsigned short* __restrict__ vT,
    unsigned short* __restrict__ aoT) {
  __shared__ __align__(16) unsigned short Qs[1024], Ks[1024], Vs[1024], Ps[1024];
  __shared__ float Ss[1024];
  const int t = blockIdx.x, h = blockIdx.y, n = blockIdx.z;
  const int lane = threadIdx.x;
  const unsigned short* qbase = qT + n * CTV + t * VV * 128 + h * HD;
  const unsigned short* kbase = kT + n * CTV + t * VV * 128 + h * HD;
  const unsigned short* vbase = vT + n * CTV + t * VV * 128 + h * HD;
  const uint4 z4 = make_uint4(0u, 0u, 0u, 0u);
  for (int i = lane; i < 128; i += 32) {        // 32 rows x 4 quads
    int w = i >> 2, qd = (i & 3) * 8;
    uint4 qv = z4, kv = z4, vv = z4;
    if (w < VV) {
      qv = *(const uint4*)(qbase + w * 128 + qd);
      kv = *(const uint4*)(kbase + w * 128 + qd);
      vv = *(const uint4*)(vbase + w * 128 + qd);
    }
    *(uint4*)(Qs + w * 32 + qd) = qv;
    *(uint4*)(Ks + w * 32 + qd) = kv;
    *(uint4*)(Vs + w * 32 + qd) = vv;
  }
  __syncthreads();
  // S[v][w] = sum_d Q[v][d] K[w][d]
#pragma unroll
  for (int mt = 0; mt < 2; ++mt) {
    v16bf a = load_A_rm(Qs + mt * 16 * 32, 32, lane);
#pragma unroll
    for (int wt = 0; wt < 2; ++wt) {
      v16bf b = load_B_cm(Ks + wt * 16 * 32, 32, lane);
      v8f s = {};
      s = wmma_bf16(a, b, s);
      int Mo = mt * 16 + ((lane >> 4) << 3);
      int cw = wt * 16 + (lane & 15);
#pragma unroll
      for (int i = 0; i < 8; ++i) Ss[(Mo + i) * 32 + cw] = s[i];
    }
  }
  __syncthreads();
  if (lane < VV) {
    const float scale = 0.17677669529663687f;   // 32^-0.5
    float mx = -3.0e38f;
    for (int w2 = 0; w2 < VV; ++w2) {
      float sv = Ss[lane * 32 + w2] * scale;
      Ss[lane * 32 + w2] = sv;
      mx = fmaxf(mx, sv);
    }
    float sum = 0.f;
    for (int w2 = 0; w2 < VV; ++w2) {
      float e = __expf(Ss[lane * 32 + w2] - mx);
      Ss[lane * 32 + w2] = e;
      sum += e;
    }
    float inv = 1.f / sum;
    for (int w2 = 0; w2 < VV; ++w2) Ps[lane * 32 + w2] = f2bf(Ss[lane * 32 + w2] * inv);
    for (int w2 = VV; w2 < 32; ++w2) Ps[lane * 32 + w2] = 0;
  } else {
    for (int w2 = 0; w2 < 32; ++w2) Ps[lane * 32 + w2] = 0;
  }
  __syncthreads();
  // OUT^T[d][v] = sum_w V^T[d][w] * P^T[w][v]  -> aoT[n][t*25+v][h*32+d]
#pragma unroll
  for (int mt = 0; mt < 2; ++mt) {
    v16bf a = load_At(Vs + mt * 16, 32, lane);        // A[d][w] = Vs[w][d0+d]
#pragma unroll
    for (int nt = 0; nt < 2; ++nt) {
      v16bf b = load_B_cm(Ps + nt * 16 * 32, 32, lane); // B[w][v] = Ps[v][w]
      v8f o = {};
      o = wmma_bf16(a, b, o);
      int Mo = (lane >> 4) << 3;
      int vcol = nt * 16 + (lane & 15);
      if (vcol < VV) {
        U16x8 u;
#pragma unroll
        for (int i = 0; i < 8; ++i) u.s[i] = f2bf(o[i]);
        *(uint4*)(aoT + n * CTV + (t * VV + vcol) * 128 + h * HD + mt * 16 + Mo) = u.q;
      }
    }
  }
}

// ---------------- wo projection + graph fuse: xfT = (Wo x ao + bo + xg)^T ----------------
__global__ void __launch_bounds__(128) k_wo(const unsigned short* __restrict__ aoT,
    const unsigned short* __restrict__ wob, const float* __restrict__ bo,
    const unsigned short* __restrict__ xgT, unsigned short* __restrict__ xfT) {
  const int col0 = blockIdx.x * 64;
  const int n = blockIdx.y;
  const int tid = threadIdx.x, lane = tid & 31, wid = tid >> 5;
  const int r0 = wid * 32;
  const unsigned short* Xn = aoT + n * CTV;
  v8f z = {};
  v8f acc[2][4];
#pragma unroll
  for (int mt = 0; mt < 2; ++mt)
#pragma unroll
    for (int nt = 0; nt < 4; ++nt) acc[mt][nt] = z;
#pragma unroll
  for (int ks = 0; ks < 128; ks += 32) {
    v16bf a0 = load_A_rm(wob + (r0) * 128 + ks, 128, lane);
    v16bf a1 = load_A_rm(wob + (r0 + 16) * 128 + ks, 128, lane);
#pragma unroll
    for (int nt = 0; nt < 4; ++nt) {
      v16bf b = load_B_cm(Xn + (col0 + nt * 16) * 128 + ks, 128, lane);
      acc[0][nt] = wmma_bf16(a0, b, acc[0][nt]);
      acc[1][nt] = wmma_bf16(a1, b, acc[1][nt]);
    }
  }
  int Mo = (lane >> 4) << 3, nc = lane & 15;
#pragma unroll
  for (int mt = 0; mt < 2; ++mt) {
    int ob = r0 + mt * 16 + Mo;
    float4 b0 = *(const float4*)(bo + ob);
    float4 b1 = *(const float4*)(bo + ob + 4);
    float bv8[8] = {b0.x, b0.y, b0.z, b0.w, b1.x, b1.y, b1.z, b1.w};
#pragma unroll
    for (int nt = 0; nt < 4; ++nt) {
      int col = col0 + nt * 16 + nc;
      int idx = n * CTV + col * 128 + ob;
      U16x8 g; g.q = *(const uint4*)(xgT + idx);
      U16x8 u;
#pragma unroll
      for (int i = 0; i < 8; ++i) u.s[i] = f2bf(acc[mt][nt][i] + bv8[i] + bf2f(g.s[i]));
      *(uint4*)(xfT + idx) = u.q;
    }
  }
}

// ---------------- tcn: OUT^T[col][o] = sum_dt xfT[col+(dt-1)*25][:] x Wdt^T; +BN +residual +PReLU ----------------
__global__ void __launch_bounds__(128) k_tcn(const unsigned short* __restrict__ xfT,
    const unsigned short* __restrict__ wt3, const float* __restrict__ btcn,
    const float* __restrict__ gam, const float* __restrict__ bet,
    const float* __restrict__ mea, const float* __restrict__ var,
    const float* __restrict__ x, const float* __restrict__ pre,
    float* __restrict__ outp) {
  const int col0 = blockIdx.x * 64;
  const int n = blockIdx.y;
  const int tid = threadIdx.x, lane = tid & 31, wid = tid >> 5;
  const int col_m = col0 + wid * 16;                 // wave's 16 output columns (M)
  const unsigned short* Xn = xfT + n * CTV;          // row indexable [col][c]
  v8f z = {};
  v8f acc[8];
#pragma unroll
  for (int ot = 0; ot < 8; ++ot) acc[ot] = z;
  for (int dt = 0; dt < 3; ++dt) {
    const int shift = (dt - 1) * VV;
    const unsigned short* Wd = wt3 + dt * 16384;     // [o][c]
#pragma unroll
    for (int ks = 0; ks < 128; ks += 32) {
      v16bf a = load_A_rm_pred(Xn + ks, 128, lane, col_m + shift, TV);
#pragma unroll
      for (int ot = 0; ot < 8; ++ot) {
        v16bf b = load_B_cm(Wd + (ot * 16) * 128 + ks, 128, lane);
        acc[ot] = wmma_bf16(a, b, acc[ot]);
      }
    }
  }
  const float pa = pre[0];
  const int Mo = (lane >> 4) << 3;
#pragma unroll
  for (int ot = 0; ot < 8; ++ot) {
    int o = ot * 16 + (lane & 15);
    float bt = btcn[o];
    float iv = gam[o] * rsqrtf(var[o] + 1e-5f);
    float mn = mea[o], be = bet[o];
    int idx = n * CTV + o * TV + col_m + Mo;
    float4 x0 = *(const float4*)(x + idx);
    float4 x1 = *(const float4*)(x + idx + 4);
    float xr[8] = {x0.x, x0.y, x0.z, x0.w, x1.x, x1.y, x1.z, x1.w};
    float yv[8];
#pragma unroll
    for (int i = 0; i < 8; ++i) {
      float y = acc[ot][i] + bt;
      y = (y - mn) * iv + be;
      y += xr[i];
      yv[i] = (y > 0.f) ? y : pa * y;
    }
    *(float4*)(outp + idx)     = make_float4(yv[0], yv[1], yv[2], yv[3]);
    *(float4*)(outp + idx + 4) = make_float4(yv[4], yv[5], yv[6], yv[7]);
  }
}

extern "C" void kernel_launch(void* const* d_in, const int* in_sizes, int n_in,
                              void* d_out, int out_size, void* d_ws, size_t ws_size,
                              hipStream_t stream) {
  (void)in_sizes; (void)n_in; (void)out_size;
  const float* x    = (const float*)d_in[0];
  const float* Aadj = (const float*)d_in[1];
  const float* Tm   = (const float*)d_in[2];
  const float* wq   = (const float*)d_in[3];
  const float* bq   = (const float*)d_in[4];
  const float* wk   = (const float*)d_in[5];
  const float* bk   = (const float*)d_in[6];
  const float* wv   = (const float*)d_in[7];
  const float* bv   = (const float*)d_in[8];
  const float* wo   = (const float*)d_in[9];
  const float* bo   = (const float*)d_in[10];
  const float* wtcn = (const float*)d_in[11];
  const float* btcn = (const float*)d_in[12];
  const float* gam  = (const float*)d_in[13];
  const float* bet  = (const float*)d_in[14];
  const float* mea  = (const float*)d_in[15];
  const float* var  = (const float*)d_in[16];
  const float* pre  = (const float*)d_in[17];
  float* outp = (float*)d_out;

  const size_t Y1C = 128ull * 64 * 128 * 32;   // 33.55M elems
  const size_t needElems = 6ull * NCTV + Y1C + 4ull * 16384 + 25ull * 4096 + 64ull * 1024 + 3ull * 16384;
  if (ws_size < needElems * sizeof(unsigned short)) return;

  unsigned short* ws = (unsigned short*)d_ws;
  size_t off = 0;
  unsigned short* xbT  = ws + off; off += NCTV;       // [n][col][c]
  unsigned short* xbT2 = ws + off; off += NCTV;       // [n][v][c][t]
  unsigned short* y1c  = ws + off; off += Y1C;        // [n][q][c][v32]; reused as aoT
  unsigned short* xgT  = ws + off; off += NCTV;       // [n][col][c]
  unsigned short* qT   = ws + off; off += NCTV;       // reused as xfT
  unsigned short* kT   = ws + off; off += NCTV;
  unsigned short* vT   = ws + off; off += NCTV;
  unsigned short* wqb  = ws + off; off += 16384;
  unsigned short* wkb  = ws + off; off += 16384;
  unsigned short* wvb  = ws + off; off += 16384;
  unsigned short* wob  = ws + off; off += 16384;
  unsigned short* tmT  = ws + off; off += 25 * 4096;
  unsigned short* aTb  = ws + off; off += 64 * 1024;
  unsigned short* wt3  = ws + off; off += 3 * 16384;

  // transposes / repacks
  { dim3 g(25, 128);  k_tr1<<<g, 128, 0, stream>>>(x, xbT); }
  { dim3 g(128, 128); k_tr2<<<g, 128, 0, stream>>>(x, xbT2); }
  k_cast<<<64, 256, 0, stream>>>(wq, wqb, 16384);
  k_cast<<<64, 256, 0, stream>>>(wk, wkb, 16384);
  k_cast<<<64, 256, 0, stream>>>(wv, wvb, 16384);
  k_cast<<<64, 256, 0, stream>>>(wo, wob, 16384);
  k_tmT<<<(25 * 4096 + 255) / 256, 256, 0, stream>>>(Tm, tmT);
  k_aT<<<(64 * 1024 + 255) / 256, 256, 0, stream>>>(Aadj, aTb);
  k_wtcn<<<(3 * 16384 + 255) / 256, 256, 0, stream>>>(wtcn, wt3);

  // graph conv
  { dim3 g(32, 128); k_gc1<<<g, 128, 0, stream>>>(xbT2, tmT, y1c); }
  { dim3 g(64, 128); k_gc2<<<g, 128, 0, stream>>>(y1c, aTb, xgT); }

  // attention pipeline
  { dim3 g(25, 128, 3); k_proj<<<g, 128, 0, stream>>>(xbT, wqb, wkb, wvb, bq, bk, bv, qT, kT, vT); }
  { dim3 g(64, 4, 128); k_attn<<<g, 32, 0, stream>>>(qT, kT, vT, y1c); }   // y1c now = aoT
  { dim3 g(25, 128);    k_wo<<<g, 128, 0, stream>>>(y1c, wob, bo, xgT, qT); } // qT now = xfT

  // tcn + BN + residual + PReLU
  { dim3 g(25, 128); k_tcn<<<g, 128, 0, stream>>>(qT, wt3, btcn, gam, bet, mea, var, x, pre, outp); }
}
